// SelfReferentialTitans_46634754900441
// MI455X (gfx1250) — compile-verified, matrix operand order checked by reading, not run on hardware
//
#include <hip/hip_runtime.h>
#include <hip/hip_bf16.h>

typedef __bf16 bf16_t;
typedef __attribute__((ext_vector_type(16))) __bf16 v16bf;
typedef __attribute__((ext_vector_type(8)))  __bf16 v8bf;
typedef __attribute__((ext_vector_type(8)))  float  v8f;
typedef __attribute__((ext_vector_type(4)))  float  v4f;

#define DIMN 512
#define BATCH 8
#define SEQ 8192
#define NTOK (BATCH * SEQ)   // 65536 rows
#define KSZ 4

union V16U { v16bf v; v8bf h[2]; };

// ---------------- weight convert (+optional transpose) fp32 -> bf16 --------
__global__ __launch_bounds__(256)
void prep_weight_kernel(const float* __restrict__ src, bf16_t* __restrict__ dst,
                        int transpose) {
  int idx = blockIdx.x * blockDim.x + threadIdx.x;   // 0 .. 512*512-1
  int r = idx >> 9, c = idx & 511;
  float v = transpose ? src[c * DIMN + r] : src[idx];
  dst[idx] = (bf16_t)v;
}

// ---------------- depthwise causal conv (fp32 math, bf16 out) --------------
__global__ __launch_bounds__(256)
void conv_kernel(const float* __restrict__ x, const float* __restrict__ w,
                 const float* __restrict__ b, bf16_t* __restrict__ out) {
  int idx = blockIdx.x * blockDim.x + threadIdx.x;   // over NTOK*DIMN
  int n = idx >> 9;            // token index = bidx*SEQ + l
  int d = idx & 511;
  int l = n & (SEQ - 1);
  float acc = b[d];
#pragma unroll
  for (int k = 0; k < KSZ; ++k) {
    int ll = l - (KSZ - 1) + k;
    if (ll >= 0)
      acc = fmaf(x[(n - (KSZ - 1) + k) * DIMN + d], w[d * KSZ + k], acc);
  }
  out[idx] = (bf16_t)acc;
}

// ---------------- fused bf16 WMMA GEMM -------------------------------------
// Per workgroup: 32 rows x 512 cols. 8 waves split N; each wave: 2 m-tiles x
// 4 n-tiles of 16x16, K-loop 16 x (K=32), register double-buffered operands.
// MODE 0: C = A(bf16) * B^T ; fuse row l2-norm ; write fp32 q
// MODE 1: C = A(fp32->bf16) * B^T ; gelu ; write bf16 h
// MODE 2: C = A(bf16) * B^T ; add fp32 residual q ; non-temporal fp32 out
template <int MODE>
__global__ __launch_bounds__(256)
void gemm_kernel(const void* __restrict__ Aptr,
                 const bf16_t* __restrict__ Bw,     // [n][k], k contiguous
                 const float* __restrict__ Qf,      // residual (MODE 2)
                 void* __restrict__ Out) {
  const int wave = threadIdx.x >> 5;
  const int lane = threadIdx.x & 31;
  const int lr = lane & 15;          // row/col within 16x16 tile
  const int lh = lane >> 4;          // K-half selector (ISA 16-bit layout)
  const int m0 = blockIdx.x * 32;    // 32 rows per workgroup
  const int n0 = wave * 64;          // 64 cols per wave

  v8f acc[2][4];
#pragma unroll
  for (int mt = 0; mt < 2; ++mt)
#pragma unroll
    for (int t = 0; t < 4; ++t)
#pragma unroll
      for (int j = 0; j < 8; ++j) acc[mt][t][j] = 0.0f;

  V16U a[2][2];   // [buffer][m-tile]
  V16U bb[2][4];  // [buffer][n-tile]

  auto loadA = [&](int buf, int kk) {
#pragma unroll
    for (int mt = 0; mt < 2; ++mt) {
      const int row = m0 + mt * 16 + lr;
      if constexpr (MODE == 1) {
        const float* A = (const float*)Aptr + row * DIMN + kk + lh * 8;
        v4f f0 = *(const v4f*)(A);
        v4f f1 = *(const v4f*)(A + 4);
        v4f f2 = *(const v4f*)(A + 16);
        v4f f3 = *(const v4f*)(A + 20);
#pragma unroll
        for (int j = 0; j < 4; ++j) {
          a[buf][mt].h[0][j]     = (bf16_t)f0[j];
          a[buf][mt].h[0][j + 4] = (bf16_t)f1[j];
          a[buf][mt].h[1][j]     = (bf16_t)f2[j];
          a[buf][mt].h[1][j + 4] = (bf16_t)f3[j];
        }
      } else {
        const bf16_t* A = (const bf16_t*)Aptr + row * DIMN + kk + lh * 8;
        a[buf][mt].h[0] = *(const v8bf*)(A);       // K = kk+lh*8 .. +7
        a[buf][mt].h[1] = *(const v8bf*)(A + 16);  // K = kk+lh*8+16 .. +23
      }
    }
  };
  auto loadB = [&](int buf, int kk) {
#pragma unroll
    for (int t = 0; t < 4; ++t) {
      const bf16_t* Bp = Bw + (n0 + t * 16 + lr) * DIMN + kk + lh * 8;
      bb[buf][t].h[0] = *(const v8bf*)(Bp);
      bb[buf][t].h[1] = *(const v8bf*)(Bp + 16);
    }
  };

  loadA(0, 0);
  loadB(0, 0);
#pragma unroll
  for (int kk = 0; kk < DIMN; kk += 32) {
    const int cur = (kk >> 5) & 1;
    const int nxt = cur ^ 1;
    if (kk + 32 < DIMN) {        // prefetch next K-slice into alternate buffer
      loadA(nxt, kk + 32);
      loadB(nxt, kk + 32);
    }
#pragma unroll
    for (int mt = 0; mt < 2; ++mt)
#pragma unroll
      for (int t = 0; t < 4; ++t)
        acc[mt][t] = __builtin_amdgcn_wmma_f32_16x16x32_bf16(
            false, a[cur][mt].v, false, bb[cur][t].v, (short)0, acc[mt][t],
            false, false);
  }

  if constexpr (MODE == 0) {
    __shared__ float rowsum[32];
    if (threadIdx.x < 32) rowsum[threadIdx.x] = 0.0f;
    __syncthreads();
#pragma unroll
    for (int mt = 0; mt < 2; ++mt)
#pragma unroll
      for (int r = 0; r < 8; ++r) {
        float s = 0.0f;
#pragma unroll
        for (int t = 0; t < 4; ++t) s += acc[mt][t][r] * acc[mt][t][r];
        atomicAdd(&rowsum[mt * 16 + r + 8 * lh], s);   // ds_add_f32
      }
    __syncthreads();
    float* q = (float*)Out;
#pragma unroll
    for (int mt = 0; mt < 2; ++mt)
#pragma unroll
      for (int r = 0; r < 8; ++r) {
        int row = mt * 16 + r + 8 * lh;
        float inv = 1.0f / fmaxf(sqrtf(rowsum[row]), 1e-12f);
#pragma unroll
        for (int t = 0; t < 4; ++t)
          q[(m0 + row) * DIMN + n0 + t * 16 + lr] = acc[mt][t][r] * inv;
      }
  } else if constexpr (MODE == 1) {
    bf16_t* h = (bf16_t*)Out;
#pragma unroll
    for (int mt = 0; mt < 2; ++mt)
#pragma unroll
      for (int r = 0; r < 8; ++r) {
        int row = m0 + mt * 16 + r + 8 * lh;
#pragma unroll
        for (int t = 0; t < 4; ++t) {
          float v = acc[mt][t][r];
          float g = 0.5f * v * (1.0f + erff(v * 0.70710678118654752f));
          h[row * DIMN + n0 + t * 16 + lr] = (bf16_t)g;
        }
      }
  } else {
    float* o = (float*)Out;
#pragma unroll
    for (int mt = 0; mt < 2; ++mt)
#pragma unroll
      for (int r = 0; r < 8; ++r) {
        int row = m0 + mt * 16 + r + 8 * lh;
#pragma unroll
        for (int t = 0; t < 4; ++t) {
          int col = n0 + t * 16 + lr;
          __builtin_nontemporal_store(acc[mt][t][r] + Qf[row * DIMN + col],
                                      &o[row * DIMN + col]);
        }
      }
  }
}

extern "C" void kernel_launch(void* const* d_in, const int* in_sizes, int n_in,
                              void* d_out, int out_size, void* d_ws, size_t ws_size,
                              hipStream_t stream) {
  const float* x  = (const float*)d_in[0];
  const float* cw = (const float*)d_in[1];
  const float* cb = (const float*)d_in[2];
  const float* qw = (const float*)d_in[3];
  const float* w1 = (const float*)d_in[4];
  const float* w2 = (const float*)d_in[5];
  float* out = (float*)d_out;

  char* ws = (char*)d_ws;
  size_t off = 0;
  bf16_t* xconv = (bf16_t*)(ws + off); off += (size_t)NTOK * DIMN * sizeof(bf16_t); // 64MB, reused as h
  float*  qbuf  = (float*) (ws + off); off += (size_t)NTOK * DIMN * sizeof(float);  // 128MB
  bf16_t* wqb   = (bf16_t*)(ws + off); off += (size_t)DIMN * DIMN * sizeof(bf16_t);
  bf16_t* w1t   = (bf16_t*)(ws + off); off += (size_t)DIMN * DIMN * sizeof(bf16_t);
  bf16_t* w2t   = (bf16_t*)(ws + off); off += (size_t)DIMN * DIMN * sizeof(bf16_t);

  dim3 blk(256);
  prep_weight_kernel<<<DIMN * DIMN / 256, blk, 0, stream>>>(qw, wqb, 0); // q_w (o,i): already [n][k]
  prep_weight_kernel<<<DIMN * DIMN / 256, blk, 0, stream>>>(w1, w1t, 1); // (i,o) -> [o][i]
  prep_weight_kernel<<<DIMN * DIMN / 256, blk, 0, stream>>>(w2, w2t, 1);

  conv_kernel<<<(NTOK * DIMN) / 256, blk, 0, stream>>>(x, cw, cb, xconv);

  // 2048 workgroups x 8 waves; each wave: 32x64 tile, K-loop 16 x (K=32)
  gemm_kernel<0><<<NTOK / 32, blk, 0, stream>>>(xconv, wqb, nullptr, qbuf);   // q = l2norm(xc @ qw^T)
  gemm_kernel<1><<<NTOK / 32, blk, 0, stream>>>(qbuf,  w1t, nullptr, xconv);  // h = gelu(q @ w1)
  gemm_kernel<2><<<NTOK / 32, blk, 0, stream>>>(xconv, w2t, qbuf,    out);    // out = q + h @ w2
}
